// PointerNetwork_87643102642501
// MI455X (gfx1250) — matrix operand order repaired and checked by reading, not compiled, not run on hardware
//
#include <hip/hip_runtime.h>
#include <hip/hip_bf16.h>
#include <math.h>

// ---------------------------------------------------------------------------
// PointerNetwork on MI455X (gfx1250, wave32, WMMA).
//   S = E = H = 512, G = 4H = 2048.
// Pipeline:
//   1) f32->bf16 converts (x, Wih, att_Wi/Wc)
//   2) WMMA GEMM (32x32 per wave, 2x2 register blocking):
//        xw = x @ Wih^T (both LSTMs)                        [bf16 -> f32 acc]
//   3) persistent single-WGP LSTM scan (enc, then dec)      [f32, LDS state]
//   4) WMMA GEMM: dproj / cproj attention projections
//   5) tiled tanh-reduction attention scores -> d_out
//   6) in-place row log_softmax on d_out
// ---------------------------------------------------------------------------

typedef __attribute__((ext_vector_type(16))) __bf16 v16bf;
typedef __attribute__((ext_vector_type(8)))  __bf16 v8bf;
typedef __attribute__((ext_vector_type(8)))  float  v8f;

#if __has_builtin(__builtin_amdgcn_tanhf)
__device__ __forceinline__ float fast_tanh(float x) {
  return __builtin_amdgcn_tanhf(x);      // native V_TANH_F32 (TRANS32)
}
#else
__device__ __forceinline__ float fast_tanh(float x) { return tanhf(x); }
#endif

__device__ __forceinline__ float sigmoidf_(float x) {
  return 1.f / (1.f + expf(-x));
}

// -------------------------- f32 -> bf16 convert ----------------------------
__global__ __launch_bounds__(256) void f32_to_bf16_kernel(
    const float* __restrict__ in, __bf16* __restrict__ out, int n) {
  int i = blockIdx.x * 256 + threadIdx.x;
  if (i < n) out[i] = (__bf16)in[i];
}

// ------------------- WMMA bf16 GEMM: C[M,N] = A[M,K] @ W[N,K]^T ------------
// One wave (32 threads) per 32x32 output tile (2x2 WMMA tiles, 4 independent
// accumulators -> consecutive v_wmma have no shared-accumulator RAW, and each
// A/B fragment is reused twice). K multiple of 32; M, N multiples of 32.
// A lane layout (16x32 bf16): lane L -> row (L&15); K-offset 16*(L>>4);
//   16 contiguous halves along K (matches B).
// B lane layout (32x16): lane L -> column (L&15) = W row; K-offset 16*(L>>4);
//   16 contiguous halves along K -> one 32B (2x b128) load.
__global__ __launch_bounds__(32) void gemm_bf16_wmma_kernel(
    const __bf16* __restrict__ A,   // [M,K] row-major
    const __bf16* __restrict__ W,   // [N,K] row-major
    float* __restrict__ C,          // [M,N]
    int M, int N, int K) {
  const int n0 = blockIdx.x * 32;
  const int m0 = blockIdx.y * 32;
  const int lane = threadIdx.x;
  const int r  = lane & 15;
  const int hi = lane >> 4;

  const __bf16* arow0 = A + (size_t)(m0 + r) * K + 16 * hi;
  const __bf16* arow1 = arow0 + (size_t)16 * K;
  const __bf16* wrow0 = W + (size_t)(n0 + r) * K + 16 * hi;
  const __bf16* wrow1 = wrow0 + (size_t)16 * K;

  v8f acc00 = {}, acc01 = {}, acc10 = {}, acc11 = {};
  for (int k = 0; k < K; k += 32) {
    v8bf a0lo = *(const v8bf*)(arow0 + k);
    v8bf a0hi = *(const v8bf*)(arow0 + k + 8);
    v8bf a1lo = *(const v8bf*)(arow1 + k);
    v8bf a1hi = *(const v8bf*)(arow1 + k + 8);
    v16bf a0 = __builtin_shufflevector(a0lo, a0hi,
        0, 1, 2, 3, 4, 5, 6, 7, 8, 9, 10, 11, 12, 13, 14, 15);
    v16bf a1 = __builtin_shufflevector(a1lo, a1hi,
        0, 1, 2, 3, 4, 5, 6, 7, 8, 9, 10, 11, 12, 13, 14, 15);
    v16bf b0 = *(const v16bf*)(wrow0 + k);
    v16bf b1 = *(const v16bf*)(wrow1 + k);
    acc00 = __builtin_amdgcn_wmma_f32_16x16x32_bf16(
        false, a0, false, b0, (short)0, acc00, false, false);
    acc01 = __builtin_amdgcn_wmma_f32_16x16x32_bf16(
        false, a0, false, b1, (short)0, acc01, false, false);
    acc10 = __builtin_amdgcn_wmma_f32_16x16x32_bf16(
        false, a1, false, b0, (short)0, acc10, false, false);
    acc11 = __builtin_amdgcn_wmma_f32_16x16x32_bf16(
        false, a1, false, b1, (short)0, acc11, false, false);
  }

  // C/D layout: VGPR i -> row m0 + mo + 8*hi + i, column n0 + no + r.
  float* c00 = C + (size_t)(m0 + 8 * hi) * N + n0 + r;
#pragma unroll
  for (int i = 0; i < 8; ++i) {
    c00[(size_t)i * N]            = acc00[i];
    c00[(size_t)i * N + 16]       = acc01[i];
    c00[(size_t)(i + 16) * N]     = acc10[i];
    c00[(size_t)(i + 16) * N + 16] = acc11[i];
  }
}

// --------------------- persistent single-WGP LSTM scan ---------------------
// One block, 1024 threads (32 waves). h and pre-activation gates live in LDS.
// Thread t computes gates t and t+1024 (dot of h with two Whh rows from L2).
__global__ __launch_bounds__(1024) void lstm_seq_kernel(
    const float* __restrict__ xw,    // [S, 2048] = x @ Wih^T (precomputed)
    const float* __restrict__ Whh,   // [2048, 512]
    const float* __restrict__ bih, const float* __restrict__ bhh,
    const float* __restrict__ h0, const float* __restrict__ c0,
    float* __restrict__ hs,          // [S, 512]
    __bf16* __restrict__ hs_bf,      // [S, 512] bf16 copy for WMMA consumers
    float* __restrict__ hN, float* __restrict__ cN, int S) {
  const int H = 512;
  __shared__ float h_s[512];
  __shared__ float gates[2048];

  const int tid = threadIdx.x;
  const int g0 = tid;
  const int g1 = tid + 1024;
  const float b0 = bih[g0] + bhh[g0];
  const float b1 = bih[g1] + bhh[g1];
  const float* __restrict__ w0 = Whh + (size_t)g0 * H;
  const float* __restrict__ w1 = Whh + (size_t)g1 * H;

  float c_reg = 0.f;
  if (tid < H) { h_s[tid] = h0[tid]; c_reg = c0[tid]; }
  __syncthreads();

  for (int s = 0; s < S; ++s) {
    float a0 = 0.f, a1 = 0.f;
#pragma unroll 8
    for (int k = 0; k < H; ++k) {
      float hk = h_s[k];                 // LDS broadcast
      a0 = fmaf(hk, w0[k], a0);
      a1 = fmaf(hk, w1[k], a1);
    }
    const float* __restrict__ xr = xw + (size_t)s * 2048;
    gates[g0] = a0 + xr[g0] + b0;
    gates[g1] = a1 + xr[g1] + b1;
    __syncthreads();
    if (tid < H) {
      float ig = sigmoidf_(gates[tid]);
      float fg = sigmoidf_(gates[512 + tid]);
      float gg = fast_tanh(gates[1024 + tid]);
      float og = sigmoidf_(gates[1536 + tid]);
      c_reg = fg * c_reg + ig * gg;
      float h = og * fast_tanh(c_reg);
      h_s[tid] = h;
      hs[(size_t)s * H + tid] = h;
      hs_bf[(size_t)s * H + tid] = (__bf16)h;
    }
    __syncthreads();
  }
  if (tid < H) { hN[tid] = h_s[tid]; cN[tid] = c_reg; }
}

// ----------------- attention scores: v . tanh(dp[t] + cp[j]) ---------------
// 16x16 (t,j) tile per block of 256 threads; H chunked by 128 through LDS.
__global__ __launch_bounds__(256) void attn_scores_kernel(
    const float* __restrict__ dproj,   // [S, H]
    const float* __restrict__ cproj,   // [S, H]
    const float* __restrict__ v,       // [H]
    float* __restrict__ scores,        // [S, S]
    int S, int H) {
  const int KC = 128;
  __shared__ float dp[16][KC];
  __shared__ float cp[16][KC];
  __shared__ float vs[KC];

  const int tj = threadIdx.x & 15;
  const int tt = threadIdx.x >> 4;
  const int j0 = blockIdx.x * 16;
  const int t0 = blockIdx.y * 16;

  float acc = 0.f;
  for (int h0 = 0; h0 < H; h0 += KC) {
    for (int i = threadIdx.x; i < 16 * KC; i += 256) {
      int rr = i / KC, cc = i % KC;
      dp[rr][cc] = dproj[(size_t)(t0 + rr) * H + h0 + cc];
      cp[rr][cc] = cproj[(size_t)(j0 + rr) * H + h0 + cc];
    }
    if (threadIdx.x < KC) vs[threadIdx.x] = v[h0 + threadIdx.x];
    __syncthreads();
#pragma unroll 4
    for (int h = 0; h < KC; ++h)
      acc += vs[h] * fast_tanh(dp[tt][h] + cp[tj][h]);
    __syncthreads();
  }
  scores[(size_t)(t0 + tt) * S + j0 + tj] = acc;
}

// ----------------------- in-place row log_softmax --------------------------
__global__ __launch_bounds__(512) void log_softmax_rows_kernel(
    float* __restrict__ x, int S) {
  __shared__ float red[512];
  const int t = blockIdx.x;
  const int j = threadIdx.x;
  float val = x[(size_t)t * S + j];

  red[j] = val;
  __syncthreads();
  for (int off = 256; off > 0; off >>= 1) {
    if (j < off) red[j] = fmaxf(red[j], red[j + off]);
    __syncthreads();
  }
  float m = red[0];
  __syncthreads();

  float e = expf(val - m);
  red[j] = e;
  __syncthreads();
  for (int off = 256; off > 0; off >>= 1) {
    if (j < off) red[j] += red[j + off];
    __syncthreads();
  }
  float lse = logf(red[0]) + m;
  x[(size_t)t * S + j] = val - lse;
}

// ---------------------------------------------------------------------------
extern "C" void kernel_launch(void* const* d_in, const int* in_sizes, int n_in,
                              void* d_out, int out_size, void* d_ws, size_t ws_size,
                              hipStream_t stream) {
  const int S = 512, E = 512, H = 512, G = 4 * H;  // G = 2048
  (void)in_sizes; (void)n_in; (void)out_size; (void)ws_size;

  const float* enc_x = (const float*)d_in[0];
  const float* dec_x = (const float*)d_in[1];
  const float* h0    = (const float*)d_in[2];
  const float* c0    = (const float*)d_in[3];
  const float* eWih  = (const float*)d_in[4];
  const float* eWhh  = (const float*)d_in[5];
  const float* ebih  = (const float*)d_in[6];
  const float* ebhh  = (const float*)d_in[7];
  const float* dWih  = (const float*)d_in[8];
  const float* dWhh  = (const float*)d_in[9];
  const float* dbih  = (const float*)d_in[10];
  const float* dbhh  = (const float*)d_in[11];
  const float* aWi   = (const float*)d_in[12];
  const float* aWc   = (const float*)d_in[13];
  const float* av    = (const float*)d_in[14];
  float* out = (float*)d_out;

  // Workspace carve-out (~20 MB total).
  char* p = (char*)d_ws;
  auto alloc = [&](size_t bytes) -> char* {
    char* r = p;
    p += (bytes + 255) & ~(size_t)255;
    return r;
  };
  float*  xw_enc  = (float*)alloc((size_t)S * G * 4);
  float*  xw_dec  = (float*)alloc((size_t)S * G * 4);
  float*  enc_hs  = (float*)alloc((size_t)S * H * 4);
  float*  dec_hs  = (float*)alloc((size_t)S * H * 4);
  float*  dproj   = (float*)alloc((size_t)S * H * 4);
  float*  cproj   = (float*)alloc((size_t)S * H * 4);
  __bf16* ex_bf   = (__bf16*)alloc((size_t)S * E * 2);
  __bf16* dx_bf   = (__bf16*)alloc((size_t)S * E * 2);
  __bf16* eWih_bf = (__bf16*)alloc((size_t)G * E * 2);
  __bf16* dWih_bf = (__bf16*)alloc((size_t)G * E * 2);
  __bf16* aWi_bf  = (__bf16*)alloc((size_t)H * H * 2);
  __bf16* aWc_bf  = (__bf16*)alloc((size_t)H * H * 2);
  __bf16* ehs_bf  = (__bf16*)alloc((size_t)S * H * 2);
  __bf16* dhs_bf  = (__bf16*)alloc((size_t)S * H * 2);
  float*  hN      = (float*)alloc((size_t)H * 4);
  float*  cN      = (float*)alloc((size_t)H * 4);

  auto cvt = [&](const float* in, __bf16* o, int n) {
    f32_to_bf16_kernel<<<(n + 255) / 256, 256, 0, stream>>>(in, o, n);
  };
  cvt(enc_x, ex_bf, S * E);
  cvt(dec_x, dx_bf, S * E);
  cvt(eWih, eWih_bf, G * E);
  cvt(dWih, dWih_bf, G * E);
  cvt(aWi, aWi_bf, H * H);
  cvt(aWc, aWc_bf, H * H);

  // Input-projection GEMMs (batch-parallel half of each LSTM).
  gemm_bf16_wmma_kernel<<<dim3(G / 32, S / 32), 32, 0, stream>>>(
      ex_bf, eWih_bf, xw_enc, S, G, E);
  gemm_bf16_wmma_kernel<<<dim3(G / 32, S / 32), 32, 0, stream>>>(
      dx_bf, dWih_bf, xw_dec, S, G, E);

  // Sequential scans (critical path): one persistent WGP each.
  lstm_seq_kernel<<<1, 1024, 0, stream>>>(
      xw_enc, eWhh, ebih, ebhh, h0, c0, enc_hs, ehs_bf, hN, cN, S);
  lstm_seq_kernel<<<1, 1024, 0, stream>>>(
      xw_dec, dWhh, dbih, dbhh, hN, cN, dec_hs, dhs_bf, hN, cN, S);

  // Attention projections.
  gemm_bf16_wmma_kernel<<<dim3(H / 32, S / 32), 32, 0, stream>>>(
      dhs_bf, aWi_bf, dproj, S, H, H);
  gemm_bf16_wmma_kernel<<<dim3(H / 32, S / 32), 32, 0, stream>>>(
      ehs_bf, aWc_bf, cproj, S, H, H);

  // Scores + log_softmax (in place on d_out).
  attn_scores_kernel<<<dim3(S / 16, S / 16), 256, 0, stream>>>(
      dproj, cproj, av, out, S, H);
  log_softmax_rows_kernel<<<S, 512, 0, stream>>>(out, S);
}
